// BranchedFF_38053410243234
// MI455X (gfx1250) — matrix operand order repaired and checked by reading, not compiled, not run on hardware
//
#include <hip/hip_runtime.h>

typedef __bf16 bf16;
typedef __attribute__((ext_vector_type(16))) __bf16 v16bf;
typedef __attribute__((ext_vector_type(8)))  __bf16 v8bf;
typedef __attribute__((ext_vector_type(4)))  __bf16 v4bf;
typedef __attribute__((ext_vector_type(8)))  float  v8f;
typedef __attribute__((ext_vector_type(4)))  float  v4f;

#define NB 32      // batch
#define NS 1024    // seq
#define ND 1024    // d_model
#define NP 4       // phases
#define NF 4096    // d_ff
#define SPL 256    // NS/NP rows per (b,p)
#define MW 32      // rows per workgroup tile
#define FC 128     // f-chunk width
// LDS strides: all == 8 (mod 16) elements -> 16B-aligned rows, <=2..4-way bank conflicts
#define XS  1032   // x tile row stride  (1024 + 8)
#define W1S 136    // W1 stage col-major stride (128 + 8)
#define W2S 40     // W2 stage col-major stride (32 + 8)
#define HS  136    // h tile row stride  (128 + 8)

#define SHUF16(lo, hi) __builtin_shufflevector((lo), (hi), \
    0,1,2,3,4,5,6,7,8,9,10,11,12,13,14,15)

__global__ __launch_bounds__(256) void branched_ff_wmma(
    const float* __restrict__ x,  const float* __restrict__ W1,
    const float* __restrict__ b1, const float* __restrict__ W2,
    const float* __restrict__ b2, float* __restrict__ out)
{
  __shared__ bf16 lds_x[MW * XS];    // 64.5 KB  x tile (bf16)
  __shared__ bf16 lds_w[ND * W2S];   // 80  KB   union: W1 stage (128*W1S) / W2 stage (1024*W2S)
  __shared__ bf16 lds_h[MW * HS];    // 8.5 KB   h chunk (bf16)

  const int tid  = threadIdx.x;
  const int wv   = tid >> 5;     // wave id 0..7
  const int lane = tid & 31;
  const int l16  = lane & 15;
  const int kh   = lane >> 4;    // lane half

  const int wg   = blockIdx.x;
  const int mt   = wg & 7;            // 8 row-tiles per (b,p)
  const int p    = (wg >> 3) & 3;
  const int bb   = wg >> 5;
  const int row0 = bb * NS + p * SPL + mt * MW;  // row into flat [B*S, D]

  // ---- stage x tile: f32 global -> bf16 LDS (coalesced b128 loads) ----
  for (int i = tid; i < MW * ND / 4; i += 256) {
    const int q = i & 255, r = i >> 8;
    const v4f xv = *(const v4f*)&x[(size_t)(row0 + r) * ND + q * 4];
    v4bf xb = { (bf16)xv[0], (bf16)xv[1], (bf16)xv[2], (bf16)xv[3] };
    *(v4bf*)&lds_x[r * XS + q * 4] = xb;
  }

  const v8f vzero = {0.f,0.f,0.f,0.f,0.f,0.f,0.f,0.f};
  v8f accY[2][8];                    // y tile: 2 M-tiles x 8 N-tiles (wave owns 128 d-cols)
#pragma unroll
  for (int t = 0; t < 2; ++t)
#pragma unroll
    for (int n = 0; n < 8; ++n) accY[t][n] = vzero;

  for (int fcI = 0; fcI < NF / FC; ++fcI) {
    const int f0 = fcI * FC;

    // ================= GEMM1: h = relu(x @ W1[:, f0:f0+FC] + b1) =================
    v8f acc1[2] = { vzero, vzero };  // wave computes h cols [wv*16, wv*16+16), M-tiles 0,1
    const float b1v = b1[p * NF + f0 + wv * 16 + l16];

    for (int ks = 0; ks < ND / 128; ++ks) {
      const int kb = ks * 128;
      __syncthreads();  // previous lds_w readers done
      // stage W1[kb..kb+128, f0..f0+FC] as column-major [f][k] (coalesced along f)
      for (int i = tid; i < 128 * FC / 4; i += 256) {
        const int q = i & 31, k = i >> 5, f = q * 4;
        const v4f wq = *(const v4f*)&W1[(size_t)(p * ND + kb + k) * NF + f0 + f];
        lds_w[(f + 0) * W1S + k] = (bf16)wq[0];
        lds_w[(f + 1) * W1S + k] = (bf16)wq[1];
        lds_w[(f + 2) * W1S + k] = (bf16)wq[2];
        lds_w[(f + 3) * W1S + k] = (bf16)wq[3];
      }
      __syncthreads();
      const bf16* pb0 = &lds_w[(wv * 16 + l16) * W1S + kh * 8];
#pragma unroll
      for (int kk = 0; kk < 4; ++kk) {          // 4 k-steps of 32 within the stage
        const int kc = kk * 32;
        // B fragment (32x16): per-lane K = {kh*8+0..7, 16+kh*8+0..7}
        const v8bf blo = *(const v8bf*)(pb0 + kc);
        const v8bf bhi = *(const v8bf*)(pb0 + kc + 16);
        const v16bf bf_ = SHUF16(blo, bhi);
#pragma unroll
        for (int t = 0; t < 2; ++t) {
          // A fragment (16x32): per-lane K = {kh*8+0..7, 16+kh*8+0..7}
          const bf16* pa = &lds_x[(t * 16 + l16) * XS + kb + kc + kh * 8];
          const v8bf alo = *(const v8bf*)pa;
          const v8bf ahi = *(const v8bf*)(pa + 16);
          const v16bf af = SHUF16(alo, ahi);
          acc1[t] = __builtin_amdgcn_wmma_f32_16x16x32_bf16(
              false, af, false, bf_, (short)0, acc1[t], false, false);
        }
      }
    }

    // bias + relu -> lds_h (C layout: VGPR r, lane -> M = r + 8*kh, N = l16)
#pragma unroll
    for (int t = 0; t < 2; ++t)
#pragma unroll
      for (int r = 0; r < 8; ++r) {
        float hv = acc1[t][r] + b1v;
        hv = hv > 0.f ? hv : 0.f;
        lds_h[(t * 16 + kh * 8 + r) * HS + wv * 16 + l16] = (bf16)hv;
      }

    // ================= GEMM2: y += h @ W2[f0:f0+FC, :] =================
    for (int ks2 = 0; ks2 < FC / 32; ++ks2) {
      const int kb2 = ks2 * 32;
      __syncthreads();  // h writes visible, lds_w free (GEMM1 done with it)
      // stage W2[f0+kb2 .. +32, :] as column-major [d][kk] (coalesced along d)
      for (int i = tid; i < 32 * ND / 4; i += 256) {
        const int q = i & 255, kk = i >> 8, d = q * 4;
        const v4f wq = *(const v4f*)&W2[(size_t)(p * NF + f0 + kb2 + kk) * ND + d];
        lds_w[(d + 0) * W2S + kk] = (bf16)wq[0];
        lds_w[(d + 1) * W2S + kk] = (bf16)wq[1];
        lds_w[(d + 2) * W2S + kk] = (bf16)wq[2];
        lds_w[(d + 3) * W2S + kk] = (bf16)wq[3];
      }
      __syncthreads();
      v16bf ah[2];
#pragma unroll
      for (int t = 0; t < 2; ++t) {
        const bf16* pa = &lds_h[(t * 16 + l16) * HS + kb2 + kh * 8];
        const v8bf alo = *(const v8bf*)pa;
        const v8bf ahi = *(const v8bf*)(pa + 16);
        ah[t] = SHUF16(alo, ahi);
      }
#pragma unroll
      for (int nb = 0; nb < 8; ++nb) {
        const bf16* pb = &lds_w[(wv * 128 + nb * 16 + l16) * W2S + kh * 8];
        const v8bf blo = *(const v8bf*)pb;
        const v8bf bhi = *(const v8bf*)(pb + 16);
        const v16bf bf_ = SHUF16(blo, bhi);
        accY[0][nb] = __builtin_amdgcn_wmma_f32_16x16x32_bf16(
            false, ah[0], false, bf_, (short)0, accY[0][nb], false, false);
        accY[1][nb] = __builtin_amdgcn_wmma_f32_16x16x32_bf16(
            false, ah[1], false, bf_, (short)0, accY[1][nb], false, false);
      }
    }
  }

  // ---- epilogue: y + b2 -> out (f32) ----
#pragma unroll
  for (int nb = 0; nb < 8; ++nb) {
    const int d = wv * 128 + nb * 16 + l16;
    const float b2v = b2[p * ND + d];
#pragma unroll
    for (int t = 0; t < 2; ++t)
#pragma unroll
      for (int r = 0; r < 8; ++r) {
        const int m = t * 16 + kh * 8 + r;
        out[(size_t)(row0 + m) * ND + d] = accY[t][nb][r] + b2v;
      }
  }
}

extern "C" void kernel_launch(void* const* d_in, const int* in_sizes, int n_in,
                              void* d_out, int out_size, void* d_ws, size_t ws_size,
                              hipStream_t stream) {
  (void)in_sizes; (void)n_in; (void)out_size; (void)d_ws; (void)ws_size;
  const float* x  = (const float*)d_in[0];
  const float* W1 = (const float*)d_in[1];
  const float* b1 = (const float*)d_in[2];
  const float* W2 = (const float*)d_in[3];
  const float* b2 = (const float*)d_in[4];
  // d_in[5] = phases: static contiguous equal partition -> implicit in indexing
  float* out = (float*)d_out;
  dim3 grid(NB * NP * (SPL / MW));   // 1024 workgroups
  dim3 block(256);                   // 8 wave32
  hipLaunchKernelGGL(branched_ff_wmma, grid, block, 0, stream,
                     x, W1, b1, W2, b2, out);
}
